// SequenceEncoder_54494545051788
// MI455X (gfx1250) — matrix operand order repaired and checked by reading, not compiled, not run on hardware
//
#include <hip/hip_runtime.h>
#include <hip/hip_bf16.h>

#define D_  1024
#define H_  16
#define HD_ 64
#define NL_ 4
#define B_  16
#define L_  1024
#define FF_ 4096
#define BL_ (B_*L_)

typedef __attribute__((ext_vector_type(16))) __bf16 v16bf;
typedef __attribute__((ext_vector_type(8)))  float  v8f;

union Frag { v16bf v; uint4 q[2]; };

__device__ __forceinline__ float silu_f(float x) { return x / (1.f + __expf(-x)); }

// LDS byte offset of a generic pointer into __shared__ memory
__device__ __forceinline__ uint32_t lds_offset(const void* p)
{
    return (uint32_t)(uintptr_t)(__attribute__((address_space(3))) const void*)p;
}

// CDNA5 async global->LDS copy, 16B per lane, tracked by ASYNCcnt
__device__ __forceinline__ void async_copy_b128(uint32_t lds_off, const void* gptr)
{
    asm volatile("global_load_async_to_lds_b128 %0, %1, off"
                 :: "v"(lds_off), "v"((uint64_t)(uintptr_t)gptr)
                 : "memory");
}

#define S_WAIT_ASYNC_4() asm volatile("s_wait_asynccnt 0x4" ::: "memory")
#define S_WAIT_ASYNC_0() asm volatile("s_wait_asynccnt 0x0" ::: "memory")

// ---------------------------------------------------------------------------
// f32 -> bf16 conversion (weight staging)
// ---------------------------------------------------------------------------
__global__ __launch_bounds__(256)
void f32_to_bf16_kernel(const float* __restrict__ src, __bf16* __restrict__ dst, int n)
{
    int i = blockIdx.x * 256 + threadIdx.x;
    if (i < n) dst[i] = (__bf16)src[i];
}

// ---------------------------------------------------------------------------
// value embed stage 1: x1 = silu(v * w_ve1 + b_ve1)  -> bf16
// ---------------------------------------------------------------------------
__global__ __launch_bounds__(256)
void embed_kernel(const float* __restrict__ values, const float* __restrict__ w1,
                  const float* __restrict__ b1, __bf16* __restrict__ XN)
{
    const int row = blockIdx.x;
    const int tid = threadIdx.x;
    const float v = values[row];
#pragma unroll
    for (int i = 0; i < 4; ++i) {
        const int dcol = tid + i * 256;
        const float x = v * w1[dcol] + b1[dcol];
        XN[(size_t)row * D_ + dcol] = (__bf16)silu_f(x);
    }
}

// ---------------------------------------------------------------------------
// RMSNorm: out = x * rsqrt(mean(x^2)+eps) * w   (bf16 out, or f32 for final)
// ---------------------------------------------------------------------------
template<bool OUT_BF16>
__global__ __launch_bounds__(256)
void rmsnorm_kernel(const float* __restrict__ X, const float* __restrict__ w,
                    __bf16* __restrict__ outb, float* __restrict__ outf)
{
    __shared__ float red[8];
    const int row = blockIdx.x;
    const int tid = threadIdx.x;
    float v[4];
    float s = 0.f;
#pragma unroll
    for (int i = 0; i < 4; ++i) {
        v[i] = X[(size_t)row * D_ + tid + i * 256];
        s += v[i] * v[i];
    }
#pragma unroll
    for (int off = 16; off; off >>= 1) s += __shfl_xor(s, off);
    if ((tid & 31) == 0) red[tid >> 5] = s;
    __syncthreads();
    float tot = 0.f;
#pragma unroll
    for (int i = 0; i < 8; ++i) tot += red[i];
    const float sc = rsqrtf(tot * (1.f / D_) + 1e-6f);
#pragma unroll
    for (int i = 0; i < 4; ++i) {
        const int dcol = tid + i * 256;
        const float y = v[i] * sc * w[dcol];
        if (OUT_BF16) outb[(size_t)row * D_ + dcol] = (__bf16)y;
        else          outf[(size_t)row * D_ + dcol] = y;
    }
}

// ---------------------------------------------------------------------------
// Shared async tile stager: ROWS x 32 bf16 tile, LDS row stride 80B
// (32 bf16 + 16B pad -> 20*row mod 64 banks distinct for 16 rows:
//  conflict-free ds_load_b128 fragment reads, 16B alignment kept).
// ---------------------------------------------------------------------------
#define TROW 80
#define TBUF (128 * TROW)           // 128-row tile = 10240 B

template<int ROWS>
__device__ __forceinline__ void stage_tile_async(const __bf16* gbase, int K, int k,
                                                 char* ldsbuf, int wid, int lane)
{
    // ROWS*4 x 16B transfers spread over 8 waves
#pragma unroll
    for (int j = 0; j < ROWS / 64; ++j) {
        const int id  = (wid * (ROWS / 64) + j) * 32 + lane;  // 0..ROWS*4-1
        const int row = id >> 2;
        const int seg = id & 3;
        const uint32_t loff = lds_offset(ldsbuf + row * TROW + seg * 16);
        async_copy_b128(loff, gbase + (size_t)row * K + k + seg * 8);
    }
}

// ---------------------------------------------------------------------------
// WMMA GEMM with async-LDS double buffering:
//   C[M,N] = A[M,K] * W[N,K]^T, bf16 operands, f32 accumulate.
// Block: 8 waves, block tile 128(M) x 128(N), K-step 32.
// Waves 4(M) x 2(N); each wave computes 32x64 (2x4 WMMA tiles).
// ---------------------------------------------------------------------------
enum { EPI_F32 = 0, EPI_PE = 1, EPI_BF16 = 2, EPI_ADDX = 3 };

template<int EPI>
__global__ __launch_bounds__(256)
void gemm_bf16_kernel(const __bf16* __restrict__ A, const __bf16* __restrict__ W,
                      float* __restrict__ Cf, __bf16* __restrict__ Cb,
                      const float* __restrict__ bias, const float* __restrict__ pe,
                      int N, int K)
{
    __shared__ char lds[2 * 2 * TBUF];   // [buf][A|W]
    const int lane = threadIdx.x & 31;
    const int wid  = threadIdx.x >> 5;
    const int wm   = wid & 3;
    const int wn   = wid >> 2;
    const int half = lane >> 4;
    const int lr   = lane & 15;

    const int mblk = blockIdx.y * 128;
    const int nblk = blockIdx.x * 128;
    const __bf16* Ablk = A + (size_t)mblk * K;
    const __bf16* Wblk = W + (size_t)nblk * K;

    const int nk = K >> 5;

    stage_tile_async<128>(Ablk, K, 0, lds,        wid, lane);
    stage_tile_async<128>(Wblk, K, 0, lds + TBUF, wid, lane);

    v8f acc[2][4] = {};

    for (int i = 0; i < nk; ++i) {
        const int cur = i & 1;
        if (i + 1 < nk) {
            char* nb = lds + (cur ^ 1) * 2 * TBUF;
            stage_tile_async<128>(Ablk, K, (i + 1) * 32, nb,        wid, lane);
            stage_tile_async<128>(Wblk, K, (i + 1) * 32, nb + TBUF, wid, lane);
            S_WAIT_ASYNC_4();
        } else {
            S_WAIT_ASYNC_0();
        }
        __syncthreads();

        const char* As = lds + cur * 2 * TBUF;
        const char* Ws = As + TBUF;

        Frag af[2], wf[4];
#pragma unroll
        for (int mi = 0; mi < 2; ++mi) {
            const char* p = As + (wm * 32 + mi * 16 + lr) * TROW + half * 32;
            af[mi].q[0] = *(const uint4*)(p);
            af[mi].q[1] = *(const uint4*)(p + 16);
        }
#pragma unroll
        for (int ni = 0; ni < 4; ++ni) {
            const char* p = Ws + (wn * 64 + ni * 16 + lr) * TROW + half * 32;
            wf[ni].q[0] = *(const uint4*)(p);
            wf[ni].q[1] = *(const uint4*)(p + 16);
        }
#pragma unroll
        for (int mi = 0; mi < 2; ++mi)
#pragma unroll
            for (int ni = 0; ni < 4; ++ni)
                acc[mi][ni] = __builtin_amdgcn_wmma_f32_16x16x32_bf16(
                    false, af[mi].v, false, wf[ni].v, (short)0, acc[mi][ni], false, false);

        __syncthreads();
    }

#pragma unroll
    for (int mi = 0; mi < 2; ++mi) {
#pragma unroll
        for (int ni = 0; ni < 4; ++ni) {
            const int col = nblk + wn * 64 + ni * 16 + lr;
#pragma unroll
            for (int r = 0; r < 8; ++r) {
                const int row = mblk + wm * 32 + mi * 16 + r + half * 8;
                float v = acc[mi][ni][r];
                if (EPI == EPI_PE)
                    Cf[(size_t)row * N + col] = v + bias[col] + pe[(size_t)(row & (L_ - 1)) * N + col];
                else if (EPI == EPI_F32)
                    Cf[(size_t)row * N + col] = v;
                else if (EPI == EPI_BF16)
                    Cb[(size_t)row * N + col] = (__bf16)v;
                else
                    Cf[(size_t)row * N + col] += v;
            }
        }
    }
}

// ---------------------------------------------------------------------------
// Fused SwiGLU dual GEMM, async-LDS double buffered:
//   H = silu(A*Wg^T) * (A*Wu^T) -> bf16
// Block: 8 waves, block tile 128(M) x 64(N); waves 4(M) x 2(N).
// Each wave: 32x32 tile for BOTH gate and up (8 WMMAs/K-step).
// LDS/buffer: A 128 rows + Wg 64 + Wu 64 = 20KB; double buffered 40KB.
// ---------------------------------------------------------------------------
#define DBUF (TBUF + 64 * TROW + 64 * TROW)   // 20480 B

__global__ __launch_bounds__(256)
void gemm_dual_swiglu_kernel(const __bf16* __restrict__ A, const __bf16* __restrict__ Wg,
                             const __bf16* __restrict__ Wu, __bf16* __restrict__ Hout,
                             int N, int K)
{
    __shared__ char lds[2 * DBUF];
    const int lane = threadIdx.x & 31;
    const int wid  = threadIdx.x >> 5;
    const int wm   = wid & 3;
    const int wn   = wid >> 2;
    const int half = lane >> 4;
    const int lr   = lane & 15;

    const int mblk = blockIdx.y * 128;
    const int nblk = blockIdx.x * 64;
    const __bf16* Ablk = A  + (size_t)mblk * K;
    const __bf16* Gblk = Wg + (size_t)nblk * K;
    const __bf16* Ublk = Wu + (size_t)nblk * K;

    const int nk = K >> 5;

    stage_tile_async<128>(Ablk, K, 0, lds,                       wid, lane);
    stage_tile_async<64> (Gblk, K, 0, lds + TBUF,                wid, lane);
    stage_tile_async<64> (Ublk, K, 0, lds + TBUF + 64 * TROW,    wid, lane);

    v8f accg[2][2] = {}, accu[2][2] = {};

    for (int i = 0; i < nk; ++i) {
        const int cur = i & 1;
        if (i + 1 < nk) {
            char* nb = lds + (cur ^ 1) * DBUF;
            const int k = (i + 1) * 32;
            stage_tile_async<128>(Ablk, K, k, nb,                    wid, lane);
            stage_tile_async<64> (Gblk, K, k, nb + TBUF,             wid, lane);
            stage_tile_async<64> (Ublk, K, k, nb + TBUF + 64 * TROW, wid, lane);
            S_WAIT_ASYNC_4();
        } else {
            S_WAIT_ASYNC_0();
        }
        __syncthreads();

        const char* As = lds + cur * DBUF;
        const char* Gs = As + TBUF;
        const char* Us = Gs + 64 * TROW;

        Frag af[2], gf[2], uf[2];
#pragma unroll
        for (int mi = 0; mi < 2; ++mi) {
            const char* p = As + (wm * 32 + mi * 16 + lr) * TROW + half * 32;
            af[mi].q[0] = *(const uint4*)(p);
            af[mi].q[1] = *(const uint4*)(p + 16);
        }
#pragma unroll
        for (int ni = 0; ni < 2; ++ni) {
            const int roff = (wn * 32 + ni * 16 + lr) * TROW + half * 32;
            gf[ni].q[0] = *(const uint4*)(Gs + roff);
            gf[ni].q[1] = *(const uint4*)(Gs + roff + 16);
            uf[ni].q[0] = *(const uint4*)(Us + roff);
            uf[ni].q[1] = *(const uint4*)(Us + roff + 16);
        }
#pragma unroll
        for (int mi = 0; mi < 2; ++mi)
#pragma unroll
            for (int ni = 0; ni < 2; ++ni) {
                accg[mi][ni] = __builtin_amdgcn_wmma_f32_16x16x32_bf16(
                    false, af[mi].v, false, gf[ni].v, (short)0, accg[mi][ni], false, false);
                accu[mi][ni] = __builtin_amdgcn_wmma_f32_16x16x32_bf16(
                    false, af[mi].v, false, uf[ni].v, (short)0, accu[mi][ni], false, false);
            }

        __syncthreads();
    }

#pragma unroll
    for (int mi = 0; mi < 2; ++mi) {
#pragma unroll
        for (int ni = 0; ni < 2; ++ni) {
            const int col = nblk + wn * 32 + ni * 16 + lr;
#pragma unroll
            for (int r = 0; r < 8; ++r) {
                const int row = mblk + wm * 32 + mi * 16 + r + half * 8;
                Hout[(size_t)row * N + col] =
                    (__bf16)(silu_f(accg[mi][ni][r]) * accu[mi][ni][r]);
            }
        }
    }
}

// ---------------------------------------------------------------------------
// V transpose: Vt[b,h,d,l] = QKV[b*L+l, 2D + h*64 + d]
// ---------------------------------------------------------------------------
__global__ __launch_bounds__(256)
void transpose_v_kernel(const __bf16* __restrict__ QKV, __bf16* __restrict__ Vt)
{
    const size_t idx = (size_t)blockIdx.x * 256 + threadIdx.x; // B*H*HD*L elems
    const int l = (int)(idx & (L_ - 1));
    const int d = (int)((idx >> 10) & (HD_ - 1));
    const int h = (int)((idx >> 16) & (H_ - 1));
    const int b = (int)(idx >> 20);
    Vt[idx] = QKV[((size_t)(b * L_ + l)) * (3 * D_) + 2 * D_ + h * HD_ + d];
}

// ---------------------------------------------------------------------------
// Flash attention, causal, one wave per 16-query tile per (b,h).
// Computes S^T = K * Q^T so softmax stats are cheap and P^T lands exactly in
// the A-fragment layout for the P*V WMMA (no cross-lane data movement).
// ---------------------------------------------------------------------------
__global__ __launch_bounds__(256)
void attn_kernel(const __bf16* __restrict__ QKV, const __bf16* __restrict__ Vt,
                 __bf16* __restrict__ O)
{
    const int lane = threadIdx.x & 31;
    const int wid  = threadIdx.x >> 5;
    const int task = blockIdx.x * 8 + wid;       // B*H*(L/16) tasks
    const int qt = task & (L_ / 16 - 1);
    const int h  = (task >> 6) & (H_ - 1);
    const int b  = task >> 10;
    const int q0 = qt * 16;
    const int half = lane >> 4;
    const int lr   = lane & 15;

    const size_t rs3 = 3 * D_;
    const __bf16* Qb = QKV + ((size_t)(b * L_ + q0)) * rs3 + h * HD_;
    const __bf16* Kb = QKV + ((size_t)(b * L_)) * rs3 + D_ + h * HD_;
    const __bf16* Vb = Vt + ((size_t)(b * H_ + h)) * HD_ * L_;

    Frag qf[2];
#pragma unroll
    for (int kk = 0; kk < 2; ++kk) {
        const __bf16* p = Qb + (size_t)lr * rs3 + kk * 32 + half * 16;
        qf[kk].q[0] = *reinterpret_cast<const uint4*>(p);
        qf[kk].q[1] = *reinterpret_cast<const uint4*>(p + 8);
    }

    v8f o[4] = {};
    float m = -INFINITY, l = 0.f;
    const float scale = 0.125f; // 1/sqrt(64)

    for (int kb = 0; kb <= q0 + 15; kb += 32) {
        v8f st[2];
#pragma unroll
        for (int t = 0; t < 2; ++t) {
            v8f s = {};
#pragma unroll
            for (int kk = 0; kk < 2; ++kk) {
                Frag kf;
                const __bf16* p = Kb + (size_t)(kb + t * 16 + lr) * rs3 + kk * 32 + half * 16;
                kf.q[0] = *reinterpret_cast<const uint4*>(p);
                kf.q[1] = *reinterpret_cast<const uint4*>(p + 8);
                s = __builtin_amdgcn_wmma_f32_16x16x32_bf16(false, kf.v, false, qf[kk].v,
                                                            (short)0, s, false, false);
            }
            st[t] = s;
        }

        const int q = q0 + lr;
        float mx = -INFINITY;
#pragma unroll
        for (int t = 0; t < 2; ++t) {
#pragma unroll
            for (int r = 0; r < 8; ++r) {
                const int key = kb + t * 16 + r + half * 8;
                float s = st[t][r] * scale;
                s = (key <= q) ? s : -INFINITY;
                st[t][r] = s;
                mx = fmaxf(mx, s);
            }
        }
        mx = fmaxf(mx, __shfl_xor(mx, 16));
        const float mnew  = fmaxf(m, mx);
        const float alpha = __expf(m - mnew);

        float rsum = 0.f;
        Frag pf;
#pragma unroll
        for (int t = 0; t < 2; ++t) {
#pragma unroll
            for (int r = 0; r < 8; ++r) {
                const float p = __expf(st[t][r] - mnew);
                rsum += p;
                pf.v[t * 8 + r] = (__bf16)p;
            }
        }
        rsum += __shfl_xor(rsum, 16);
        l = l * alpha + rsum;
        m = mnew;

#pragma unroll
        for (int r = 0; r < 8; ++r) {
            const float ar = __shfl(alpha, r + half * 8);
            o[0][r] *= ar; o[1][r] *= ar; o[2][r] *= ar; o[3][r] *= ar;
        }

#pragma unroll
        for (int dc = 0; dc < 4; ++dc) {
            Frag vf;
            const __bf16* p = Vb + (size_t)(dc * 16 + lr) * L_ + kb + half * 8;
            vf.q[0] = *reinterpret_cast<const uint4*>(p);
            vf.q[1] = *reinterpret_cast<const uint4*>(p + 16);
            o[dc] = __builtin_amdgcn_wmma_f32_16x16x32_bf16(false, pf.v, false, vf.v,
                                                            (short)0, o[dc], false, false);
        }
    }

#pragma unroll
    for (int r = 0; r < 8; ++r) {
        const float li  = __shfl(l, r + half * 8);
        const float inv = 1.f / li;
        const int row = q0 + r + half * 8;
        __bf16* op = O + ((size_t)(b * L_ + row)) * D_ + h * HD_ + lr;
        op[0]  = (__bf16)(o[0][r] * inv);
        op[16] = (__bf16)(o[1][r] * inv);
        op[32] = (__bf16)(o[2][r] * inv);
        op[48] = (__bf16)(o[3][r] * inv);
    }
}

// ---------------------------------------------------------------------------
// mean pool over L: z[b,d] = mean_l xfinal[b,l,d]
// ---------------------------------------------------------------------------
__global__ __launch_bounds__(256)
void meanpool_kernel(const float* __restrict__ Xn, float* __restrict__ z)
{
    const int dcol = blockIdx.x * 256 + threadIdx.x;
    const int b = blockIdx.y;
    float s = 0.f;
    for (int l = 0; l < L_; ++l)
        s += Xn[((size_t)(b * L_ + l)) * D_ + dcol];
    z[(size_t)b * D_ + dcol] = s * (1.f / L_);
}

// ---------------------------------------------------------------------------
extern "C" void kernel_launch(void* const* d_in, const int* in_sizes, int n_in,
                              void* d_out, int out_size, void* d_ws, size_t ws_size,
                              hipStream_t stream)
{
    const float* values  = (const float*)d_in[0];
    const float* w_ve1   = (const float*)d_in[1];
    const float* b_ve1   = (const float*)d_in[2];
    const float* w_ve2   = (const float*)d_in[3];
    const float* b_ve2   = (const float*)d_in[4];
    const float* pe      = (const float*)d_in[5];
    const float* norm1_w = (const float*)d_in[6];
    const float* qkv_w   = (const float*)d_in[7];
    const float* out_w   = (const float*)d_in[8];
    const float* norm2_w = (const float*)d_in[9];
    const float* gate_w  = (const float*)d_in[10];
    const float* up_w    = (const float*)d_in[11];
    const float* down_w  = (const float*)d_in[12];
    const float* fnorm_w = (const float*)d_in[13];

    char* ws = (char*)d_ws;
    size_t off = 0;
    auto alloc = [&](size_t bytes) {
        void* p = ws + off;
        off += (bytes + 255) & ~(size_t)255;
        return p;
    };
    float*  X   = (float*)alloc((size_t)BL_ * D_ * 4);
    __bf16* XN  = (__bf16*)alloc((size_t)BL_ * D_ * 2);
    __bf16* QKV = (__bf16*)alloc((size_t)BL_ * 3 * D_ * 2);
    __bf16* Vt  = (__bf16*)alloc((size_t)BL_ * D_ * 2);
    __bf16* Ob  = (__bf16*)alloc((size_t)BL_ * D_ * 2);
    __bf16* Hff = (__bf16*)alloc((size_t)BL_ * FF_ * 2);
    __bf16* Wg  = (__bf16*)alloc((size_t)D_ * FF_ * 2);
    __bf16* Wu  = (__bf16*)alloc((size_t)D_ * FF_ * 2);
    __bf16* Wst = (__bf16*)alloc((size_t)D_ * FF_ * 2);

    float* z    = (float*)d_out;
    float* Xout = z + (size_t)B_ * D_;

    const dim3 blk(256);
    auto cvt = [&](const float* s, __bf16* d, int n) {
        f32_to_bf16_kernel<<<dim3((n + 255) / 256), blk, 0, stream>>>(s, d, n);
    };

    // ---- embed: x = silu(v*w1+b1) @ w2^T + b2 + pe
    cvt(w_ve2, Wst, D_ * D_);
    embed_kernel<<<dim3(BL_), blk, 0, stream>>>(values, w_ve1, b_ve1, XN);
    gemm_bf16_kernel<EPI_PE><<<dim3(D_ / 128, BL_ / 128), blk, 0, stream>>>(
        XN, Wst, X, nullptr, b_ve2, pe, D_, D_);

    for (int i = 0; i < NL_; ++i) {
        // attention block
        rmsnorm_kernel<true><<<dim3(BL_), blk, 0, stream>>>(X, norm1_w + (size_t)i * D_, XN, nullptr);
        cvt(qkv_w + (size_t)i * 3 * D_ * D_, Wst, 3 * D_ * D_);
        gemm_bf16_kernel<EPI_BF16><<<dim3(3 * D_ / 128, BL_ / 128), blk, 0, stream>>>(
            XN, Wst, nullptr, QKV, nullptr, nullptr, 3 * D_, D_);
        transpose_v_kernel<<<dim3((B_ * H_ * HD_ * L_) / 256), blk, 0, stream>>>(QKV, Vt);
        attn_kernel<<<dim3(B_ * H_ * (L_ / 16) / 8), blk, 0, stream>>>(QKV, Vt, Ob);
        cvt(out_w + (size_t)i * D_ * D_, Wst, D_ * D_);
        gemm_bf16_kernel<EPI_ADDX><<<dim3(D_ / 128, BL_ / 128), blk, 0, stream>>>(
            Ob, Wst, X, nullptr, nullptr, nullptr, D_, D_);

        // SwiGLU FFN block
        rmsnorm_kernel<true><<<dim3(BL_), blk, 0, stream>>>(X, norm2_w + (size_t)i * D_, XN, nullptr);
        cvt(gate_w + (size_t)i * D_ * FF_, Wg, D_ * FF_);
        cvt(up_w   + (size_t)i * D_ * FF_, Wu, D_ * FF_);
        gemm_dual_swiglu_kernel<<<dim3(FF_ / 64, BL_ / 128), blk, 0, stream>>>(
            XN, Wg, Wu, Hff, FF_, D_);
        cvt(down_w + (size_t)i * D_ * FF_, Wst, D_ * FF_);
        gemm_bf16_kernel<EPI_ADDX><<<dim3(D_ / 128, BL_ / 128), blk, 0, stream>>>(
            Hff, Wst, X, nullptr, nullptr, nullptr, D_, FF_);
    }

    // ---- final norm + outputs
    rmsnorm_kernel<false><<<dim3(BL_), blk, 0, stream>>>(X, fnorm_w, nullptr, Xout);
    meanpool_kernel<<<dim3(D_ / 256, B_), blk, 0, stream>>>(Xout, z);

    (void)in_sizes; (void)n_in; (void)out_size; (void)ws_size;
}
